// JitLSTMEncoder_46978352284260
// MI455X (gfx1250) — matrix operand order repaired and checked by reading, not compiled, not run on hardware
//
#include <hip/hip_runtime.h>

// ---------------------------------------------------------------------------
// Problem constants (match reference)
// ---------------------------------------------------------------------------
#define N_B     16
#define T_S     512
#define F_INW   320
#define IN_PROJ 1024
#define H_SZ    1024
#define P_SZ    512
#define G4      4096
#define L_N     4
#define OUT_SZ  512
#define EPS_LN  1e-5f

typedef __bf16 bf16_t;
typedef __attribute__((ext_vector_type(16))) __bf16 v16bf;
typedef __attribute__((ext_vector_type(8)))  __bf16 v8bf;
typedef __attribute__((ext_vector_type(8)))  float  v8f;

// Recurrent scratch layout (bytes, per direction)
#define REC_OFF_H     0            // h state: 16x512 bf16          (16384)
#define REC_OFF_C     16384        // c state: 16x1024 f32          (65536)
#define REC_OFF_GACC  81920        // raw gates: 16x4096 f32        (262144)
#define REC_OFF_SO    344064       // sigmoid(o): 16x1024 f32       (65536)
#define REC_OFF_HINT  409600       // pre-proj h: 16x1024 bf16      (32768)
#define REC_OFF_GST   442368       // gate LN partials 16wg*16*2 f32(2048)
#define REC_OFF_CST   444416       // c LN partials                  (2048)
#define REC_OFF_BAR   446464       // barrier count/gen              (256)
#define REC_DIR_BYTES 458752       // 448 KB, padded

// ---------------------------------------------------------------------------
// Helpers
// ---------------------------------------------------------------------------
__device__ inline v8f zero_v8f() {
  v8f z;
#pragma unroll
  for (int i = 0; i < 8; ++i) z[i] = 0.f;
  return z;
}

__device__ inline v8f wmma_bf16(v16bf a, v16bf b, v8f c) {
  // D = A(16x32) * B(32x16) + C, f32 accumulate
  return __builtin_amdgcn_wmma_f32_16x16x32_bf16(false, a, false, b,
                                                 (short)0, c, false, false);
}

// A fragment: 16x32 bf16 tile at (m0, k0) of row-major A (lda elements/row).
// ISA layout: lane L (<16): M=L, K = {0..7 , 16..23}; lane L+16: K = {8..15, 24..31}
__device__ inline v16bf load_a_frag(const bf16_t* __restrict__ A, int lda,
                                    int m0, int k0) {
  const int lane = threadIdx.x & 31;
  const int row  = m0 + (lane & 15);
  const int kb   = k0 + ((lane >> 4) << 3);           // +0 or +8
  const bf16_t* p = A + (size_t)row * lda + kb;
  v8bf lo = *(const v8bf*)p;                          // K = kb .. kb+7
  v8bf hi = *(const v8bf*)(p + 16);                   // K = kb+16 .. kb+23
  v16bf r;
#pragma unroll
  for (int i = 0; i < 8; ++i) { r[i] = lo[i]; r[i + 8] = hi[i]; }
  return r;
}

// B fragment for C = A * W^T with W row-major (Nout x K):
// lane L (<16): column N=n0+L, K = {0..7, 16..23}; lane L+16: K = {8..15, 24..31}
__device__ inline v16bf load_b_frag(const bf16_t* __restrict__ W, int ldb,
                                    int n0, int k0) {
  const int lane = threadIdx.x & 31;
  const int col  = n0 + (lane & 15);
  const int kb   = k0 + ((lane >> 4) << 3);           // +0 or +8
  const bf16_t* p = W + (size_t)col * ldb + kb;
  v8bf lo = *(const v8bf*)p;
  v8bf hi = *(const v8bf*)(p + 16);
  v16bf r;
#pragma unroll
  for (int i = 0; i < 8; ++i) { r[i] = lo[i]; r[i + 8] = hi[i]; }
  return r;
}

__device__ inline float sigmoidf_(float x) { return 1.f / (1.f + expf(-x)); }

// Cross-workgroup barrier for the 16 WGs of one direction.
// Preferred path: CDNA5 cluster user barrier (S_BARRIER_SIGNAL/WAIT -3).
// Fallback (not launched as a cluster -> cluster_id==0): global sense barrier.
__device__ inline void dir_barrier(unsigned* bar, int nwg) {
  __syncthreads();
  __threadfence();
  if (__builtin_amdgcn_cluster_id_x() != 0) {
    if ((threadIdx.x >> 5) == 0) {          // one wave per WG signals
      asm volatile("s_barrier_signal -3" ::: "memory");
    }
    asm volatile("s_barrier_wait -3" ::: "memory");  // all waves wait
  } else if (threadIdx.x == 0) {
    unsigned my_gen = __hip_atomic_load(&bar[1], __ATOMIC_ACQUIRE,
                                        __HIP_MEMORY_SCOPE_AGENT);
    unsigned prev = __hip_atomic_fetch_add(&bar[0], 1u, __ATOMIC_ACQ_REL,
                                           __HIP_MEMORY_SCOPE_AGENT);
    if (prev == (unsigned)nwg - 1u) {
      __hip_atomic_store(&bar[0], 0u, __ATOMIC_RELAXED,
                         __HIP_MEMORY_SCOPE_AGENT);
      __hip_atomic_fetch_add(&bar[1], 1u, __ATOMIC_RELEASE,
                             __HIP_MEMORY_SCOPE_AGENT);
    } else {
      while (__hip_atomic_load(&bar[1], __ATOMIC_ACQUIRE,
                               __HIP_MEMORY_SCOPE_AGENT) == my_gen) {
        __builtin_amdgcn_s_sleep(8);
      }
    }
  }
  __syncthreads();
}

// ---------------------------------------------------------------------------
// f32 -> bf16 convert
// ---------------------------------------------------------------------------
__global__ __launch_bounds__(256) void cvt_f32_bf16_k(const float* __restrict__ in,
                                                      bf16_t* __restrict__ out,
                                                      long n) {
  long i = (long)blockIdx.x * blockDim.x + threadIdx.x;
  const long stride = (long)gridDim.x * blockDim.x;
  for (; i < n; i += stride) out[i] = (bf16_t)in[i];
}

__global__ __launch_bounds__(256) void zero_u32_k(unsigned* __restrict__ p, long n) {
  long i = (long)blockIdx.x * blockDim.x + threadIdx.x;
  const long stride = (long)gridDim.x * blockDim.x;
  for (; i < n; i += stride) p[i] = 0u;
}

// ---------------------------------------------------------------------------
// Generic WMMA GEMM: C(MxN) = A(MxK) * W^T (W is NxK, row-major) + bias
// Block: 256 threads = 8 waves, tile 128(M) x 64(N); wave = 32x32 (2x2 WMMA)
// M % 128 == 0, N % 64 == 0, K % 32 == 0 (true for all call sites)
// ---------------------------------------------------------------------------
__global__ __launch_bounds__(256)
void gemm_bf16_wmma(const bf16_t* __restrict__ A, const bf16_t* __restrict__ W,
                    const float* __restrict__ bias,
                    float* __restrict__ Cf, bf16_t* __restrict__ Cb,
                    int M, int K, int Nout, int relu) {
  const int wave = threadIdx.x >> 5;
  const int lane = threadIdx.x & 31;
  const int m0 = blockIdx.y * 128 + (wave >> 1) * 32;
  const int n0 = blockIdx.x * 64  + (wave & 1) * 32;

  v8f acc[2][2];
#pragma unroll
  for (int i = 0; i < 2; ++i)
#pragma unroll
    for (int j = 0; j < 2; ++j) acc[i][j] = zero_v8f();

  for (int k = 0; k < K; k += 32) {
    if (k + 128 < K) { // CDNA5 prefetch hint -> global_prefetch_b8
      __builtin_prefetch(W + (size_t)(n0 + (lane & 15)) * K + k + 128, 0, 1);
      __builtin_prefetch(W + (size_t)(n0 + 16 + (lane & 15)) * K + k + 128, 0, 1);
    }
    v16bf a0 = load_a_frag(A, K, m0, k);
    v16bf a1 = load_a_frag(A, K, m0 + 16, k);
    v16bf b0 = load_b_frag(W, K, n0, k);
    v16bf b1 = load_b_frag(W, K, n0 + 16, k);
    acc[0][0] = wmma_bf16(a0, b0, acc[0][0]);
    acc[0][1] = wmma_bf16(a0, b1, acc[0][1]);
    acc[1][0] = wmma_bf16(a1, b0, acc[1][0]);
    acc[1][1] = wmma_bf16(a1, b1, acc[1][1]);
  }

  const int colL = lane & 15;
  const int rb   = (lane >> 4) * 8;   // C layout: lane>=16 holds rows M=8..15
#pragma unroll
  for (int ti = 0; ti < 2; ++ti) {
#pragma unroll
    for (int tj = 0; tj < 2; ++tj) {
      const int col = n0 + tj * 16 + colL;
      const float bv = bias ? bias[col] : 0.f;
#pragma unroll
      for (int r = 0; r < 8; ++r) {
        const int row = m0 + ti * 16 + rb + r;
        float v = acc[ti][tj][r] + bv;
        if (relu) v = fmaxf(v, 0.f);
        if (Cf) Cf[(size_t)row * Nout + col] = v;
        if (Cb) Cb[(size_t)row * Nout + col] = (bf16_t)v;
      }
    }
  }
}

// ---------------------------------------------------------------------------
// Row LayerNorm (in place, f32): one block per row
// ---------------------------------------------------------------------------
__global__ __launch_bounds__(256)
void ln_rows_inplace(float* __restrict__ X, const float* __restrict__ g,
                     const float* __restrict__ b, int C) {
  __shared__ float ssum[256], ssq[256];
  float* x = X + (size_t)blockIdx.x * C;
  float s = 0.f, s2 = 0.f;
  for (int c = threadIdx.x; c < C; c += 256) { float v = x[c]; s += v; s2 += v * v; }
  ssum[threadIdx.x] = s; ssq[threadIdx.x] = s2;
  __syncthreads();
  for (int off = 128; off > 0; off >>= 1) {
    if ((int)threadIdx.x < off) {
      ssum[threadIdx.x] += ssum[threadIdx.x + off];
      ssq[threadIdx.x]  += ssq[threadIdx.x + off];
    }
    __syncthreads();
  }
  const float mu  = ssum[0] / (float)C;
  const float var = ssq[0] / (float)C - mu * mu;
  const float inv = rsqrtf(var + EPS_LN);
  for (int c = threadIdx.x; c < C; c += 256)
    x[c] = (x[c] - mu) * inv * g[c] + b[c];
}

// ---------------------------------------------------------------------------
// Persistent bidirectional LSTM layer. Grid = 32 WGs (16 per direction, one
// cluster's worth each), 256 threads (8 waves). Runs all T steps:
//   P1: gates = h @ w_hh^T + b_hh      (WMMA, 256 gate cols per WG)
//   P2: LN(gates)+ig -> i,f,g,o -> c   (64 h-cols per WG)
//   P3: LN(c) -> h_int = sig(o)*tanh   (64 h-cols per WG)
//   P4: h = h_int @ w_p^T              (WMMA, 32 cols per WG) -> state + output
// ---------------------------------------------------------------------------
__global__ __launch_bounds__(256)
void lstm_rec_kernel(const float* __restrict__ ig_all,
                     const bf16_t* __restrict__ whh_all,
                     const float* __restrict__ bhh_all,
                     const float* __restrict__ ghg_all,
                     const float* __restrict__ bhg_all,
                     const float* __restrict__ gc_all,
                     const float* __restrict__ bc_all,
                     const bf16_t* __restrict__ wp_all,
                     bf16_t* __restrict__ xout,
                     unsigned char* __restrict__ rec_all) {
  const int dir  = blockIdx.x >> 4;
  const int wg   = blockIdx.x & 15;
  const int tid  = threadIdx.x;
  const int wave = tid >> 5;
  const int lane = tid & 31;

  const float*  ig  = ig_all  + (size_t)dir * N_B * T_S * G4;
  const bf16_t* whh = whh_all + (size_t)dir * G4 * P_SZ;
  const float*  bhh = bhh_all + dir * G4;
  const float*  ghg = ghg_all + dir * G4;
  const float*  bhg = bhg_all + dir * G4;
  const float*  gc  = gc_all  + dir * H_SZ;
  const float*  bc  = bc_all  + dir * H_SZ;
  const bf16_t* wp  = wp_all  + (size_t)dir * P_SZ * H_SZ;
  unsigned char* rec = rec_all + (size_t)dir * REC_DIR_BYTES;

  bf16_t*   h       = (bf16_t*)(rec + REC_OFF_H);
  float*    c_state = (float*) (rec + REC_OFF_C);
  float*    gacc    = (float*) (rec + REC_OFF_GACC);
  float*    o_gate  = (float*) (rec + REC_OFF_SO);
  bf16_t*   hint    = (bf16_t*)(rec + REC_OFF_HINT);
  float*    gst     = (float*) (rec + REC_OFF_GST);
  float*    cstp    = (float*) (rec + REC_OFF_CST);
  unsigned* bar     = (unsigned*)(rec + REC_OFF_BAR);

  __shared__ float sm0[16], sm1[16], sm2[16], sm3[16];

  for (int t = 0; t < T_S; ++t) {
    const int tt = dir ? (T_S - 1 - t) : t;

    // ---- Phase 1: raw gates GEMM for cols [wg*256, wg*256+256) ----
    {
      const int nbase = wg * 256 + wave * 32;
      v8f acc0 = zero_v8f(), acc1 = zero_v8f();
      for (int k = 0; k < P_SZ; k += 32) {
        v16bf a  = load_a_frag(h, P_SZ, 0, k);
        v16bf b0 = load_b_frag(whh, P_SZ, nbase, k);
        v16bf b1 = load_b_frag(whh, P_SZ, nbase + 16, k);
        acc0 = wmma_bf16(a, b0, acc0);
        acc1 = wmma_bf16(a, b1, acc1);
      }
      if (tid < 16) { sm0[tid] = 0.f; sm1[tid] = 0.f; }
      __syncthreads();
      const int colL = lane & 15;
      const int rb   = (lane >> 4) * 8;
#pragma unroll
      for (int tj = 0; tj < 2; ++tj) {
        const v8f acc = tj ? acc1 : acc0;
        const int col = nbase + tj * 16 + colL;
        const float bv = bhh[col];
#pragma unroll
        for (int r = 0; r < 8; ++r) {
          const int row = rb + r;
          float v = acc[r] + bv;
          gacc[row * G4 + col] = v;
          atomicAdd(&sm0[row], v);
          atomicAdd(&sm1[row], v * v);
        }
      }
      __syncthreads();
      if (tid < 16) {
        gst[(wg * 16 + tid) * 2 + 0] = sm0[tid];
        gst[(wg * 16 + tid) * 2 + 1] = sm1[tid];
      }
    }
    dir_barrier(bar, 16);

    // ---- Phase 2: LN(gates)+ig, cell update for h-cols [wg*64, +64) ----
    {
      if (tid < 16) {
        float S = 0.f, S2 = 0.f;
        for (int w = 0; w < 16; ++w) {
          S  += gst[(w * 16 + tid) * 2 + 0];
          S2 += gst[(w * 16 + tid) * 2 + 1];
        }
        const float mu  = S * (1.f / (float)G4);
        const float var = S2 * (1.f / (float)G4) - mu * mu;
        sm0[tid] = mu;
        sm1[tid] = rsqrtf(var + EPS_LN);
        sm2[tid] = 0.f;   // c-sum partial
        sm3[tid] = 0.f;   // c-sumsq partial
      }
      __syncthreads();
      for (int e = tid; e < 1024; e += 256) {
        const int row = e >> 6;
        const int j   = (wg << 6) + (e & 63);
        const float mu = sm0[row], inv = sm1[row];
        const float* ga    = gacc + row * G4;
        const float* igrow = ig + ((size_t)row * T_S + tt) * G4;
        float iv = (ga[j]            - mu) * inv * ghg[j]            + bhg[j]            + igrow[j];
        float fv = (ga[H_SZ + j]     - mu) * inv * ghg[H_SZ + j]     + bhg[H_SZ + j]     + igrow[H_SZ + j];
        float gv = (ga[2 * H_SZ + j] - mu) * inv * ghg[2 * H_SZ + j] + bhg[2 * H_SZ + j] + igrow[2 * H_SZ + j];
        float ov = (ga[3 * H_SZ + j] - mu) * inv * ghg[3 * H_SZ + j] + bhg[3 * H_SZ + j] + igrow[3 * H_SZ + j];
        const float si = sigmoidf_(iv), sf = sigmoidf_(fv);
        const float tg = tanhf(gv),     so = sigmoidf_(ov);
        const float cn = sf * c_state[row * H_SZ + j] + si * tg;
        c_state[row * H_SZ + j] = cn;
        o_gate[row * H_SZ + j]  = so;
        atomicAdd(&sm2[row], cn);
        atomicAdd(&sm3[row], cn * cn);
      }
      __syncthreads();
      if (tid < 16) {
        cstp[(wg * 16 + tid) * 2 + 0] = sm2[tid];
        cstp[(wg * 16 + tid) * 2 + 1] = sm3[tid];
      }
    }
    dir_barrier(bar, 16);

    // ---- Phase 3: LN(c), h_int = sig(o)*tanh(LN(c)) ----
    {
      if (tid < 16) {
        float S = 0.f, S2 = 0.f;
        for (int w = 0; w < 16; ++w) {
          S  += cstp[(w * 16 + tid) * 2 + 0];
          S2 += cstp[(w * 16 + tid) * 2 + 1];
        }
        const float mu  = S * (1.f / (float)H_SZ);
        const float var = S2 * (1.f / (float)H_SZ) - mu * mu;
        sm0[tid] = mu;
        sm1[tid] = rsqrtf(var + EPS_LN);
      }
      __syncthreads();
      for (int e = tid; e < 1024; e += 256) {
        const int row = e >> 6;
        const int j   = (wg << 6) + (e & 63);
        const float cv  = c_state[row * H_SZ + j];
        const float lnc = (cv - sm0[row]) * sm1[row] * gc[j] + bc[j];
        const float hv  = o_gate[row * H_SZ + j] * tanhf(lnc);
        hint[row * H_SZ + j] = (bf16_t)hv;
      }
    }
    dir_barrier(bar, 16);

    // ---- Phase 4: projection h = h_int @ w_p^T, cols [wg*32, +32) ----
    {
      if (wave < 2) {
        const int nbase = wg * 32 + wave * 16;
        v8f acc = zero_v8f();
        for (int k = 0; k < H_SZ; k += 32) {
          v16bf a = load_a_frag(hint, H_SZ, 0, k);
          v16bf b = load_b_frag(wp, H_SZ, nbase, k);
          acc = wmma_bf16(a, b, acc);
        }
        const int colL = lane & 15;
        const int rb   = (lane >> 4) * 8;
        const int col  = nbase + colL;
#pragma unroll
        for (int r = 0; r < 8; ++r) {
          const int row = rb + r;
          const bf16_t hv = (bf16_t)acc[r];
          h[row * P_SZ + col] = hv;
          xout[((size_t)row * T_S + tt) * (2 * P_SZ) + dir * P_SZ + col] = hv;
        }
      }
    }
    dir_barrier(bar, 16);
  }
}

// ---------------------------------------------------------------------------
// Host orchestration
// ---------------------------------------------------------------------------
extern "C" void kernel_launch(void* const* d_in, const int* in_sizes, int n_in,
                              void* d_out, int out_size, void* d_ws, size_t ws_size,
                              hipStream_t stream) {
  (void)in_sizes; (void)n_in; (void)out_size; (void)ws_size;
  const float* inp     = (const float*)d_in[0];
  const int*   inp_len = (const int*)  d_in[1];
  const float* proj_w  = (const float*)d_in[2];
  const float* proj_b  = (const float*)d_in[3];
  const float* w_ih    = (const float*)d_in[4];
  const float* b_ih    = (const float*)d_in[5];
  const float* w_hh    = (const float*)d_in[6];
  const float* b_hh    = (const float*)d_in[7];
  const float* ln_ig_g = (const float*)d_in[8];
  const float* ln_ig_b = (const float*)d_in[9];
  const float* ln_hg_g = (const float*)d_in[10];
  const float* ln_hg_b = (const float*)d_in[11];
  const float* ln_c_g  = (const float*)d_in[12];
  const float* ln_c_b  = (const float*)d_in[13];
  const float* w_proj  = (const float*)d_in[14];
  const float* out_w   = (const float*)d_in[15];
  const float* out_b   = (const float*)d_in[16];

  size_t off = 0;
  auto arena = [&](size_t bytes) -> void* {
    void* p = (char*)d_ws + off;
    off += (bytes + 255) & ~(size_t)255;
    return p;
  };
  const size_t ROWS = (size_t)N_B * T_S;  // 8192

  bf16_t* bf_projw = (bf16_t*)arena((size_t)IN_PROJ * F_INW * 2);
  bf16_t* bf_wih   = (bf16_t*)arena((size_t)L_N * 2 * G4 * IN_PROJ * 2);
  bf16_t* bf_whh   = (bf16_t*)arena((size_t)L_N * 2 * G4 * P_SZ * 2);
  bf16_t* bf_wp    = (bf16_t*)arena((size_t)L_N * 2 * P_SZ * H_SZ * 2);
  bf16_t* bf_outw  = (bf16_t*)arena((size_t)OUT_SZ * IN_PROJ * 2);
  bf16_t* bf_inp   = (bf16_t*)arena(ROWS * F_INW * 2);
  bf16_t* xA       = (bf16_t*)arena(ROWS * IN_PROJ * 2);
  bf16_t* xB       = (bf16_t*)arena(ROWS * IN_PROJ * 2);
  float*  igbuf    = (float*) arena((size_t)2 * ROWS * G4 * 4);
  unsigned char* rec = (unsigned char*)arena((size_t)2 * REC_DIR_BYTES);

  // Weight / activation conversion to bf16 (keeps all weights L2-resident)
  cvt_f32_bf16_k<<<512,  256, 0, stream>>>(proj_w, bf_projw, (long)IN_PROJ * F_INW);
  cvt_f32_bf16_k<<<4096, 256, 0, stream>>>(w_ih,   bf_wih,   (long)L_N * 2 * G4 * IN_PROJ);
  cvt_f32_bf16_k<<<4096, 256, 0, stream>>>(w_hh,   bf_whh,   (long)L_N * 2 * G4 * P_SZ);
  cvt_f32_bf16_k<<<2048, 256, 0, stream>>>(w_proj, bf_wp,    (long)L_N * 2 * P_SZ * H_SZ);
  cvt_f32_bf16_k<<<512,  256, 0, stream>>>(out_w,  bf_outw,  (long)OUT_SZ * IN_PROJ);
  cvt_f32_bf16_k<<<2048, 256, 0, stream>>>(inp,    bf_inp,   (long)ROWS * F_INW);

  // x0 = relu(inp @ proj_w^T + proj_b)  -> bf16
  {
    dim3 g(IN_PROJ / 64, (unsigned)(ROWS / 128));
    gemm_bf16_wmma<<<g, 256, 0, stream>>>(bf_inp, bf_projw, proj_b,
                                          nullptr, xA, (int)ROWS, F_INW, IN_PROJ, 1);
  }

  bf16_t* xin = xA;
  bf16_t* xnx = xB;
  for (int l = 0; l < L_N; ++l) {
    // ig[d] = LN(x @ w_ih^T + b_ih)  (parallel over all n,t)
    for (int d = 0; d < 2; ++d) {
      const int ld = l * 2 + d;
      float* igd = igbuf + (size_t)d * ROWS * G4;
      dim3 g(G4 / 64, (unsigned)(ROWS / 128));
      gemm_bf16_wmma<<<g, 256, 0, stream>>>(xin, bf_wih + (size_t)ld * G4 * IN_PROJ,
                                            b_ih + (size_t)ld * G4,
                                            igd, nullptr, (int)ROWS, IN_PROJ, G4, 0);
      ln_rows_inplace<<<(unsigned)ROWS, 256, 0, stream>>>(
          igd, ln_ig_g + (size_t)ld * G4, ln_ig_b + (size_t)ld * G4, G4);
    }
    // zero h/c state, scratch and barrier words for both directions
    zero_u32_k<<<256, 256, 0, stream>>>((unsigned*)rec, (long)(2 * REC_DIR_BYTES / 4));
    // persistent recurrent kernel: 16 WGs per direction
    lstm_rec_kernel<<<32, 256, 0, stream>>>(
        igbuf,
        bf_whh + (size_t)l * 2 * G4 * P_SZ,
        b_hh    + (size_t)l * 2 * G4,
        ln_hg_g + (size_t)l * 2 * G4,  ln_hg_b + (size_t)l * 2 * G4,
        ln_c_g  + (size_t)l * 2 * H_SZ, ln_c_b + (size_t)l * 2 * H_SZ,
        bf_wp   + (size_t)l * 2 * P_SZ * H_SZ,
        xnx, rec);
    bf16_t* tmp = xin; xin = xnx; xnx = tmp;
  }

  // out = x @ out_w^T + out_b  -> f32 straight into d_out
  {
    dim3 g(OUT_SZ / 64, (unsigned)(ROWS / 128));
    gemm_bf16_wmma<<<g, 256, 0, stream>>>(xin, bf_outw, out_b,
                                          (float*)d_out, nullptr,
                                          (int)ROWS, IN_PROJ, OUT_SZ, 0);
  }
  // second tuple element: inp_len passed through (bit-copy into output tail)
  hipMemcpyAsync((char*)d_out + ROWS * OUT_SZ * sizeof(float), inp_len,
                 N_B * sizeof(int), hipMemcpyDeviceToDevice, stream);
}
